// ST_BLOCK_0_47158740910282
// MI455X (gfx1250) — matrix operand order repaired and checked by reading, not compiled, not run on hardware
//
#include <hip/hip_runtime.h>
#include <cstdint>
#include <cstddef>

// ST_BLOCK for MI455X (gfx1250, wave32, WMMA).
// Heavy contractions -> v_wmma_f32_16x16x32_f16 (f32 accumulate), LDS-staged tiles,
// 128-bit global loads, transposed B staging so fragment builds are wide ds reads.
// Precision-critical elementwise / softmax / LayerNorm -> fp32 VALU.

#define B_  16
#define C_  32
#define N_  1024
#define L_  24
#define CO_ 64
#define KT_ 3

typedef __attribute__((ext_vector_type(16))) _Float16 v16h;
typedef __attribute__((ext_vector_type(8)))  _Float16 v8h;
typedef __attribute__((ext_vector_type(8)))  float    v8f;
typedef _Float16 h16;

// ---------------- workspace layout (bytes) ----------------
constexpr size_t ALN(size_t x) { return (x + 255) & ~size_t(255); }
constexpr size_t OFF_XIN   = 0;                                            // f32 [B,CO,N,L] (later "pre", in-place)
constexpr size_t OFF_GCN   = ALN(OFF_XIN  + sizeof(float) * B_*CO_*N_*L_); // f32 [B,CO,N,L]
constexpr size_t OFF_XTAT  = ALN(OFF_GCN  + sizeof(float) * B_*CO_*N_*L_); // f16 [B,C,N,L] (x_TAt)
constexpr size_t OFF_RAWS  = ALN(OFF_XTAT + sizeof(h16)   * B_*C_*N_*L_);  // f16 [B,N,N]   (satt raw)
constexpr size_t OFF_SH    = ALN(OFF_RAWS + sizeof(h16)   * (size_t)B_*N_*N_); // f16 [B,N,N] (S_coef)
constexpr size_t OFF_Y     = ALN(OFF_SH   + sizeof(h16)   * (size_t)B_*N_*N_); // f16 [2,B,C,N,L]
constexpr size_t OFF_SUPH  = ALN(OFF_Y    + sizeof(h16)   * 2ull*B_*C_*N_*L_); // f16 [N,N]
constexpr size_t OFF_SVH   = ALN(OFF_SUPH + sizeof(h16)   * (size_t)N_*N_);    // f16 [N,N]
constexpr size_t OFF_T2H   = ALN(OFF_SVH  + sizeof(h16)   * (size_t)N_*N_);    // f16 [N,N]
constexpr size_t OFF_F1T   = ALN(OFF_T2H  + sizeof(h16)   * (size_t)N_*N_);    // f32 [B,L,N]
constexpr size_t OFF_F2T   = ALN(OFF_F1T  + sizeof(float) * B_*L_*N_);     // f32 [B,C,L]
constexpr size_t OFF_GT    = ALN(OFF_F2T  + sizeof(float) * B_*C_*L_);     // f32 [B,L,C]
constexpr size_t OFF_F1S   = ALN(OFF_GT   + sizeof(float) * B_*L_*C_);     // f32 [B,N,L]
constexpr size_t OFF_F2S   = ALN(OFF_F1S  + sizeof(float) * B_*N_*L_);     // f32 [B,C,N]
constexpr size_t OFF_GS    = ALN(OFF_F2S  + sizeof(float) * B_*C_*N_);     // f32 [B,N,C]
constexpr size_t OFF_CMAX  = ALN(OFF_GS   + sizeof(float) * B_*N_*C_);     // f32 [B,N]
constexpr size_t OFF_PART  = ALN(OFF_CMAX + sizeof(float) * B_*N_);        // f32 [B,96,2]
constexpr size_t OFF_STATS = ALN(OFF_PART + sizeof(float) * B_*96*2);      // f32 [B,2]
// total ~342 MB; assumed <= ws_size

#define SHUF16(a, b) __builtin_shufflevector((a), (b), 0,1,2,3,4,5,6,7,8,9,10,11,12,13,14,15)

__device__ __forceinline__ float block_reduce_256(float v, float* sm) {
  int t = threadIdx.x;
  sm[t] = v; __syncthreads();
  for (int s = 128; s > 0; s >>= 1) { if (t < s) sm[t] += sm[t + s]; __syncthreads(); }
  float r = sm[0]; __syncthreads();
  return r;
}

// ---------------- elementwise f32 -> f16 ----------------
__global__ void k_cvt_h(const float* __restrict__ a, h16* __restrict__ o, int n) {
  int i = blockIdx.x * 256 + threadIdx.x;
  if (i < n) o[i] = (h16)a[i];
}

// ---------------- conv1 (1x1): x_input[b,o,n,l] ----------------
__global__ void k_conv1(const float* __restrict__ x, const float* __restrict__ w,
                        const float* __restrict__ bias, float* __restrict__ xin) {
  int b = blockIdx.x >> 10, n = blockIdx.x & (N_ - 1);
  __shared__ float xs[C_][L_];
  __shared__ float ws[CO_ * C_];
  for (int i = threadIdx.x; i < CO_ * C_; i += 256) ws[i] = w[i];
  for (int i = threadIdx.x; i < C_ * L_; i += 256) {
    int c = i / L_, l = i % L_;
    xs[c][l] = x[(((size_t)b * C_ + c) * N_ + n) * L_ + l];
  }
  __syncthreads();
  for (int i = threadIdx.x; i < CO_ * L_; i += 256) {
    int o = i / L_, l = i % L_;
    float acc = bias[o];
    #pragma unroll
    for (int c = 0; c < C_; ++c) acc += ws[o * C_ + c] * xs[c][l];
    xin[(((size_t)b * CO_ + o) * N_ + n) * L_ + l] = acc;
  }
}

// ---------------- TATT small stages ----------------
__global__ void k_tf1(const float* __restrict__ x, const float* __restrict__ w1,
                      float* __restrict__ f1) { // f1[b,l,n] = sum_c x[b,c,n,l]*w1[c]
  int idx = blockIdx.x * 256 + threadIdx.x;          // B*L*N
  int b = idx / (L_ * N_), r = idx % (L_ * N_), l = r / N_, n = r % N_;
  float acc = 0.f;
  #pragma unroll
  for (int c = 0; c < C_; ++c) acc += x[(((size_t)b * C_ + c) * N_ + n) * L_ + l] * w1[c];
  f1[((size_t)b * L_ + l) * N_ + n] = acc;
}
__global__ void k_tf2(const float* __restrict__ x, const float* __restrict__ w2,
                      float* __restrict__ f2) { // f2[b,c,l] = sum_n x[b,c,n,l]*w2[n]
  int idx = blockIdx.x * 256 + threadIdx.x;          // B*C*L
  int b = idx / (C_ * L_), r = idx % (C_ * L_), c = r / L_, l = r % L_;
  float acc = 0.f;
  for (int n = 0; n < N_; ++n) acc += x[(((size_t)b * C_ + c) * N_ + n) * L_ + l] * w2[n];
  f2[((size_t)b * C_ + c) * L_ + l] = acc;
}
__global__ void k_tg(const float* __restrict__ f1, const float* __restrict__ tw,
                     float* __restrict__ g) { // g[b,l,c] = sum_n f1[b,l,n]*tw[n,c]
  int idx = blockIdx.x * 256 + threadIdx.x;          // B*L*C
  int b = idx / (L_ * C_), r = idx % (L_ * C_), l = r / C_, c = r % C_;
  float acc = 0.f;
  for (int n = 0; n < N_; ++n) acc += f1[((size_t)b * L_ + l) * N_ + n] * tw[n * C_ + c];
  g[((size_t)b * L_ + l) * C_ + c] = acc;
}
// raw -> logits -> col-max-subtract -> row softmax -> transposed store (one block per b)
__global__ void k_tatt(const float* __restrict__ g, const float* __restrict__ f2,
                       const float* __restrict__ tb, const float* __restrict__ tv,
                       float* __restrict__ Treg) {
  int b = blockIdx.x;
  __shared__ float raw[L_][L_], lg[L_][L_], cm[L_], rs[L_];
  for (int i = threadIdx.x; i < L_ * L_; i += 256) {
    int l = i / L_, q = i % L_;
    float a = tb[l * L_ + q];
    #pragma unroll
    for (int c = 0; c < C_; ++c)
      a += g[((size_t)b * L_ + l) * C_ + c] * f2[((size_t)b * C_ + c) * L_ + q];
    raw[l][q] = 1.f / (1.f + __expf(-a));
  }
  __syncthreads();
  for (int i = threadIdx.x; i < L_ * L_; i += 256) {
    int m = i / L_, q = i % L_;
    float a = 0.f;
    #pragma unroll
    for (int k = 0; k < L_; ++k) a += tv[m * L_ + k] * raw[k][q];
    lg[m][q] = a;
  }
  __syncthreads();
  for (int q = threadIdx.x; q < L_; q += 256) {
    float mx = lg[0][q];
    for (int m = 1; m < L_; ++m) mx = fmaxf(mx, lg[m][q]);
    cm[q] = mx;
  }
  __syncthreads();
  for (int i = threadIdx.x; i < L_ * L_; i += 256) {
    int m = i / L_, q = i % L_;
    lg[m][q] = __expf(lg[m][q] - cm[q]);
  }
  __syncthreads();
  for (int m = threadIdx.x; m < L_; m += 256) {
    float s = 0.f;
    for (int q = 0; q < L_; ++q) s += lg[m][q];
    rs[m] = s;
  }
  __syncthreads();
  for (int i = threadIdx.x; i < L_ * L_; i += 256) {
    int m = i / L_, q = i % L_;
    Treg[(size_t)b * L_ * L_ + q * L_ + m] = lg[m][q] / rs[m]; // transposed
  }
}

// ---------------- SATT small stages ----------------
__global__ void k_sf1(const float* __restrict__ x, const float* __restrict__ w1,
                      float* __restrict__ f1) { // f1[b,n,l] = sum_c x[b,c,n,l]*w1[c]
  int idx = blockIdx.x * 256 + threadIdx.x;          // B*N*L
  int b = idx / (N_ * L_), r = idx % (N_ * L_), n = r / L_, l = r % L_;
  float acc = 0.f;
  #pragma unroll
  for (int c = 0; c < C_; ++c) acc += x[(((size_t)b * C_ + c) * N_ + n) * L_ + l] * w1[c];
  f1[((size_t)b * N_ + n) * L_ + l] = acc;
}
__global__ void k_sf2(const float* __restrict__ x, const float* __restrict__ w2,
                      float* __restrict__ f2) { // f2[b,c,m] = sum_l x[b,c,m,l]*w2[l]
  int idx = blockIdx.x * 256 + threadIdx.x;          // B*C*N
  int b = idx / (C_ * N_), r = idx % (C_ * N_), c = r / N_, m = r % N_;
  float acc = 0.f;
  #pragma unroll
  for (int l = 0; l < L_; ++l) acc += x[(((size_t)b * C_ + c) * N_ + m) * L_ + l] * w2[l];
  f2[((size_t)b * C_ + c) * N_ + m] = acc;
}
__global__ void k_sg(const float* __restrict__ f1, const float* __restrict__ sw,
                     float* __restrict__ g) { // g[b,n,c] = sum_l f1[b,n,l]*sw[l,c]
  int idx = blockIdx.x * 256 + threadIdx.x;          // B*N*C
  int b = idx / (N_ * C_), r = idx % (N_ * C_), n = r / C_, c = r % C_;
  float acc = 0.f;
  #pragma unroll
  for (int l = 0; l < L_; ++l) acc += f1[((size_t)b * N_ + n) * L_ + l] * sw[l * C_ + c];
  g[((size_t)b * N_ + n) * C_ + c] = acc;
}
__global__ void k_sraw(const float* __restrict__ g, const float* __restrict__ f2,
                       const float* __restrict__ sb, h16* __restrict__ raw) {
  int idx = blockIdx.x * 256 + threadIdx.x;          // B*N*N (m fastest -> coalesced)
  int b = idx >> 20, r = idx & ((N_ * N_) - 1), n = r >> 10, m = r & (N_ - 1);
  float acc = sb[n * N_ + m];
  #pragma unroll
  for (int c = 0; c < C_; ++c)
    acc += g[((size_t)b * N_ + n) * C_ + c] * f2[((size_t)b * C_ + c) * N_ + m];
  raw[idx] = (h16)(1.f / (1.f + __expf(-acc)));
}

// =====================================================================
// WMMA GEMM kernels: 256 thr = 8 waves, workgroup tile 64(M) x 64(N),
// wave tile 32(M) x 16(N) (two stacked 16x16 tiles share one B fragment).
// A staged row-major As[64][40]; B staged TRANSPOSED Bs[col][k] so each
// lane's fragment is two contiguous 16B ds reads. 40-half row pad keeps
// 16B alignment of all fragment chunks.
// =====================================================================

// logits[b,m,q] = sum_k s_v[m,k] * raw[b,k,q]
__global__ void __launch_bounds__(256)
k_logits_wmma(const h16* __restrict__ svh, const h16* __restrict__ raw,
              float* __restrict__ logits) {
  __shared__ alignas(16) h16 As[64][40];
  __shared__ alignas(16) h16 Bs[64][40];
  int tid = threadIdx.x;
  int wave = tid >> 5, lane = tid & 31, hf = lane >> 4, lm = lane & 15;
  int wn = wave & 3, wm = wave >> 2;
  int m0 = blockIdx.x * 64, n0 = blockIdx.y * 64, b = blockIdx.z;
  const h16* Bg = raw + (size_t)b * N_ * N_;
  int arow = tid >> 2, ack = (tid & 3) * 8;
  int brow = tid >> 3, bck = (tid & 7) * 8;
  v8f acc0 = {}, acc1 = {};
  for (int k0 = 0; k0 < N_; k0 += 32) {
    *reinterpret_cast<v8h*>(&As[arow][ack]) =
        *reinterpret_cast<const v8h*>(&svh[(size_t)(m0 + arow) * N_ + k0 + ack]);
    v8h bchunk = *reinterpret_cast<const v8h*>(&Bg[(size_t)(k0 + brow) * N_ + n0 + bck]);
    #pragma unroll
    for (int j = 0; j < 8; ++j) Bs[bck + j][brow] = bchunk[j];
    __syncthreads();
    v8h b0 = *reinterpret_cast<const v8h*>(&Bs[wn * 16 + lm][hf * 8]);
    v8h b1 = *reinterpret_cast<const v8h*>(&Bs[wn * 16 + lm][16 + hf * 8]);
    v16h bf = SHUF16(b0, b1);
    int r0 = wm * 32 + lm;
    v16h af0 = SHUF16(*reinterpret_cast<const v8h*>(&As[r0][hf * 8]),
                      *reinterpret_cast<const v8h*>(&As[r0][16 + hf * 8]));
    v16h af1 = SHUF16(*reinterpret_cast<const v8h*>(&As[r0 + 16][hf * 8]),
                      *reinterpret_cast<const v8h*>(&As[r0 + 16][16 + hf * 8]));
    acc0 = __builtin_amdgcn_wmma_f32_16x16x32_f16(false, af0, false, bf, (short)0, acc0, false, false);
    acc1 = __builtin_amdgcn_wmma_f32_16x16x32_f16(false, af1, false, bf, (short)0, acc1, false, false);
    __syncthreads();
  }
  float* out = logits + (size_t)b * N_ * N_;
  int q = n0 + wn * 16 + lm;
  #pragma unroll
  for (int r = 0; r < 8; ++r) {
    out[(size_t)(m0 + wm * 32 + r + 8 * hf) * N_ + q]      = acc0[r];
    out[(size_t)(m0 + wm * 32 + 16 + r + 8 * hf) * N_ + q] = acc1[r];
  }
}

// T2 = 2*supports@supports - I   (f16 out)
__global__ void __launch_bounds__(256)
k_cheb2_wmma(const h16* __restrict__ suph, h16* __restrict__ t2h) {
  __shared__ alignas(16) h16 As[64][40];
  __shared__ alignas(16) h16 Bs[64][40];
  int tid = threadIdx.x;
  int wave = tid >> 5, lane = tid & 31, hf = lane >> 4, lm = lane & 15;
  int wn = wave & 3, wm = wave >> 2;
  int m0 = blockIdx.x * 64, n0 = blockIdx.y * 64;
  int arow = tid >> 2, ack = (tid & 3) * 8;
  int brow = tid >> 3, bck = (tid & 7) * 8;
  v8f acc0 = {}, acc1 = {};
  for (int k0 = 0; k0 < N_; k0 += 32) {
    *reinterpret_cast<v8h*>(&As[arow][ack]) =
        *reinterpret_cast<const v8h*>(&suph[(size_t)(m0 + arow) * N_ + k0 + ack]);
    v8h bchunk = *reinterpret_cast<const v8h*>(&suph[(size_t)(k0 + brow) * N_ + n0 + bck]);
    #pragma unroll
    for (int j = 0; j < 8; ++j) Bs[bck + j][brow] = bchunk[j];
    __syncthreads();
    v8h b0 = *reinterpret_cast<const v8h*>(&Bs[wn * 16 + lm][hf * 8]);
    v8h b1 = *reinterpret_cast<const v8h*>(&Bs[wn * 16 + lm][16 + hf * 8]);
    v16h bf = SHUF16(b0, b1);
    int r0 = wm * 32 + lm;
    v16h af0 = SHUF16(*reinterpret_cast<const v8h*>(&As[r0][hf * 8]),
                      *reinterpret_cast<const v8h*>(&As[r0][16 + hf * 8]));
    v16h af1 = SHUF16(*reinterpret_cast<const v8h*>(&As[r0 + 16][hf * 8]),
                      *reinterpret_cast<const v8h*>(&As[r0 + 16][16 + hf * 8]));
    acc0 = __builtin_amdgcn_wmma_f32_16x16x32_f16(false, af0, false, bf, (short)0, acc0, false, false);
    acc1 = __builtin_amdgcn_wmma_f32_16x16x32_f16(false, af1, false, bf, (short)0, acc1, false, false);
    __syncthreads();
  }
  int q = n0 + wn * 16 + lm;
  #pragma unroll
  for (int r = 0; r < 8; ++r) {
    int ma = m0 + wm * 32 + r + 8 * hf;
    int mb = ma + 16;
    t2h[(size_t)ma * N_ + q] = (h16)(2.f * acc0[r] - (ma == q ? 1.f : 0.f));
    t2h[(size_t)mb * N_ + q] = (h16)(2.f * acc1[r] - (mb == q ? 1.f : 0.f));
  }
}

// xg GEMM (k=1,2): Y[kp,b,c,q,l] = sum_n (S[b,q,n]*Tk[q,n]) * xT[b,c,n,l]
// GEMM cols j = c*L + l (768); 8-col chunks never cross a c boundary (24 = 3*8),
// so B staging stays 16B-aligned contiguous global loads.
__global__ void __launch_bounds__(256)
k_xg_wmma(const h16* __restrict__ Sh, const h16* __restrict__ suph,
          const h16* __restrict__ t2h, const h16* __restrict__ xT,
          h16* __restrict__ Y) {
  __shared__ alignas(16) h16 As[64][40];
  __shared__ alignas(16) h16 Bs[64][40];
  int tid = threadIdx.x;
  int wave = tid >> 5, lane = tid & 31, hf = lane >> 4, lm = lane & 15;
  int wn = wave & 3, wm = wave >> 2;
  int m0 = blockIdx.x * 64;            // q tile
  int n0 = blockIdx.y * 64;            // (c,l) col tile
  int b = blockIdx.z >> 1, kp = blockIdx.z & 1;
  const h16* Tk = kp ? t2h : suph;
  const h16* Sb = Sh + (size_t)b * N_ * N_;
  int arow = tid >> 2, ack = (tid & 3) * 8;
  int brow = tid >> 3, bck = (tid & 7) * 8;
  int bc = (n0 + bck) / L_, bl = (n0 + bck) % L_;     // chunk's (c, l0)
  v8f acc0 = {}, acc1 = {};
  for (int n0k = 0; n0k < N_; n0k += 32) {
    size_t arowg = (size_t)(m0 + arow) * N_ + n0k + ack;
    v8h s = *reinterpret_cast<const v8h*>(&Sb[arowg]);
    v8h t = *reinterpret_cast<const v8h*>(&Tk[arowg]);
    *reinterpret_cast<v8h*>(&As[arow][ack]) = s * t;   // v_pk_mul_f16 pair
    v8h bchunk = *reinterpret_cast<const v8h*>(
        &xT[(((size_t)b * C_ + bc) * N_ + (n0k + brow)) * L_ + bl]);
    #pragma unroll
    for (int j = 0; j < 8; ++j) Bs[bck + j][brow] = bchunk[j];
    __syncthreads();
    v8h b0 = *reinterpret_cast<const v8h*>(&Bs[wn * 16 + lm][hf * 8]);
    v8h b1 = *reinterpret_cast<const v8h*>(&Bs[wn * 16 + lm][16 + hf * 8]);
    v16h bf = SHUF16(b0, b1);
    int r0 = wm * 32 + lm;
    v16h af0 = SHUF16(*reinterpret_cast<const v8h*>(&As[r0][hf * 8]),
                      *reinterpret_cast<const v8h*>(&As[r0][16 + hf * 8]));
    v16h af1 = SHUF16(*reinterpret_cast<const v8h*>(&As[r0 + 16][hf * 8]),
                      *reinterpret_cast<const v8h*>(&As[r0 + 16][16 + hf * 8]));
    acc0 = __builtin_amdgcn_wmma_f32_16x16x32_f16(false, af0, false, bf, (short)0, acc0, false, false);
    acc1 = __builtin_amdgcn_wmma_f32_16x16x32_f16(false, af1, false, bf, (short)0, acc1, false, false);
    __syncthreads();
  }
  int col = n0 + wn * 16 + lm, c = col / L_, l = col % L_;
  size_t obase = (((size_t)kp * B_ + b) * C_ + c) * N_;
  #pragma unroll
  for (int r = 0; r < 8; ++r) {
    int qa = m0 + wm * 32 + r + 8 * hf;
    Y[(obase + qa) * L_ + l]        = (h16)acc0[r];
    Y[(obase + qa + 16) * L_ + l]   = (h16)acc1[r];
  }
}

// ---------------- S softmax: col-max then row softmax (in-place in d_out) ----------------
__global__ void k_colmax(const float* __restrict__ logits, float* __restrict__ cmax) {
  int idx = blockIdx.x * 256 + threadIdx.x;          // B*N
  int b = idx >> 10, q = idx & (N_ - 1);
  const float* p = logits + (size_t)b * N_ * N_ + q;
  float mx = p[0];
  for (int m = 1; m < N_; ++m) mx = fmaxf(mx, p[(size_t)m * N_]);
  cmax[idx] = mx;
}
__global__ void k_softmax_s(float* __restrict__ S, const float* __restrict__ cmax,
                            h16* __restrict__ Sh) {
  int b = blockIdx.x >> 10, m = blockIdx.x & (N_ - 1);
  __shared__ float sm[256];
  float* row = S + (size_t)b * N_ * N_ + (size_t)m * N_;
  const float* cm = cmax + (size_t)b * N_;
  float s = 0.f;
  for (int q = threadIdx.x; q < N_; q += 256) s += __expf(row[q] - cm[q]);
  float tot = block_reduce_256(s, sm);
  float inv = 1.f / tot;
  for (int q = threadIdx.x; q < N_; q += 256) {
    float e = __expf(row[q] - cm[q]) * inv;
    row[q] = e;
    Sh[(size_t)b * N_ * N_ + (size_t)m * N_ + q] = (h16)e;
  }
}

// ---------------- x_TAt[b,c,n,q] = sum_l x[b,c,n,l] * T_coef[b,l,q]  (f16 out) -------
__global__ void k_xtat(const float* __restrict__ x, const float* __restrict__ Treg,
                       h16* __restrict__ xT) {
  int b = blockIdx.x >> 10, n = blockIdx.x & (N_ - 1);
  __shared__ float xs[C_][L_], tc[L_][L_];
  for (int i = threadIdx.x; i < C_ * L_; i += 256) {
    int c = i / L_, l = i % L_;
    xs[c][l] = x[(((size_t)b * C_ + c) * N_ + n) * L_ + l];
  }
  for (int i = threadIdx.x; i < L_ * L_; i += 256)
    tc[i / L_][i % L_] = Treg[(size_t)b * L_ * L_ + i];
  __syncthreads();
  for (int i = threadIdx.x; i < C_ * L_; i += 256) {
    int c = i / L_, q = i % L_;
    float acc = 0.f;
    #pragma unroll
    for (int l = 0; l < L_; ++l) acc += xs[c][l] * tc[l][q];
    xT[(((size_t)b * C_ + c) * N_ + n) * L_ + q] = (h16)acc;
  }
}

// ---------------- gcn combine: relu( b + sum_{k,c} w[o,kC+c] * term_k[c,q,l] ) ------
__global__ void k_gcn_combine(const h16* __restrict__ xT, const h16* __restrict__ Y,
                              const float* __restrict__ S, const float* __restrict__ gw,
                              const float* __restrict__ gb, float* __restrict__ gcn) {
  int b = blockIdx.x >> 10, q = blockIdx.x & (N_ - 1);
  __shared__ float sx[3][C_][L_];
  __shared__ float sw[CO_ * 3 * C_];
  float sd = S[(size_t)b * N_ * N_ + (size_t)q * N_ + q];   // diag(S_coef)
  for (int i = threadIdx.x; i < CO_ * 3 * C_; i += 256) sw[i] = gw[i];
  for (int i = threadIdx.x; i < C_ * L_; i += 256) {
    int c = i / L_, l = i % L_;
    size_t base = (((size_t)b * C_ + c) * N_ + q) * L_ + l;
    sx[0][c][l] = (float)xT[base] * sd;                       // k=0 (identity term)
    sx[1][c][l] = (float)Y[base];                             // k=1 (supports)
    sx[2][c][l] = (float)Y[(size_t)B_ * C_ * N_ * L_ + base]; // k=2 (T2)
  }
  __syncthreads();
  for (int i = threadIdx.x; i < CO_ * L_; i += 256) {
    int o = i / L_, l = i % L_;
    float acc = gb[o];
    #pragma unroll
    for (int k = 0; k < 3; ++k)
      #pragma unroll
      for (int c = 0; c < C_; ++c)
        acc += sw[o * (3 * C_) + k * C_ + c] * sx[k][c][l];
    gcn[(((size_t)b * CO_ + o) * N_ + q) * L_ + l] = fmaxf(acc, 0.f);
  }
}

// ---------------- time conv (1,3) pad(0,1) + residual + relu (pre overwrites xin) ----
__global__ void k_timeconv(const float* __restrict__ gcn, const float* __restrict__ tw,
                           const float* __restrict__ tb, float* __restrict__ pre) {
  int b = blockIdx.x >> 10, n = blockIdx.x & (N_ - 1);
  __shared__ float g[CO_][L_ + 1];
  __shared__ float sw[CO_ * CO_ * KT_];
  for (int i = threadIdx.x; i < CO_ * CO_ * KT_; i += 256) sw[i] = tw[i];
  for (int i = threadIdx.x; i < CO_ * L_; i += 256) {
    int o = i / L_, l = i % L_;
    g[o][l] = gcn[(((size_t)b * CO_ + o) * N_ + n) * L_ + l];
  }
  __syncthreads();
  for (int i = threadIdx.x; i < CO_ * L_; i += 256) {
    int o = i / L_, l = i % L_;
    float acc = tb[o];
    #pragma unroll
    for (int o2 = 0; o2 < CO_; ++o2) {
      #pragma unroll
      for (int dt = 0; dt < KT_; ++dt) {
        int ll = l + dt - 1;
        if (ll >= 0 && ll < L_) acc += sw[(o * CO_ + o2) * KT_ + dt] * g[o2][ll];
      }
    }
    size_t idx = (((size_t)b * CO_ + o) * N_ + n) * L_ + l;
    pre[idx] = fmaxf(acc + pre[idx], 0.f);   // residual with x_input held in same buffer
  }
}

// ---------------- LayerNorm (deterministic two-stage reduction) ----------------
__global__ void k_ln_partial(const float* __restrict__ pre, float* __restrict__ part) {
  int b = blockIdx.x / 96, blk = blockIdx.x % 96;
  __shared__ float sm[256];
  size_t base = (size_t)b * (CO_ * N_ * L_) + (size_t)blk * 16384;
  float s = 0.f, q = 0.f;
  for (int i = threadIdx.x; i < 16384; i += 256) {
    float v = pre[base + i];
    s += v; q += v * v;
  }
  float ts = block_reduce_256(s, sm);
  float tq = block_reduce_256(q, sm);
  if (threadIdx.x == 0) {
    part[(size_t)(b * 96 + blk) * 2 + 0] = ts;
    part[(size_t)(b * 96 + blk) * 2 + 1] = tq;
  }
}
__global__ void k_ln_final(const float* __restrict__ part, float* __restrict__ stats) {
  int b = blockIdx.x;
  __shared__ float s0[128], s1[128];
  float a = 0.f, q = 0.f;
  for (int i = threadIdx.x; i < 96; i += 128) {
    a += part[(size_t)(b * 96 + i) * 2 + 0];
    q += part[(size_t)(b * 96 + i) * 2 + 1];
  }
  s0[threadIdx.x] = a; s1[threadIdx.x] = q; __syncthreads();
  for (int s = 64; s > 0; s >>= 1) {
    if (threadIdx.x < s) { s0[threadIdx.x] += s0[threadIdx.x + s]; s1[threadIdx.x] += s1[threadIdx.x + s]; }
    __syncthreads();
  }
  if (threadIdx.x == 0) {
    float cnt = (float)(CO_ * N_ * L_);
    float mu = s0[0] / cnt;
    float var = s1[0] / cnt - mu * mu;
    stats[b * 2 + 0] = mu;
    stats[b * 2 + 1] = rsqrtf(var + 1e-5f);
  }
}
__global__ void k_ln_apply(const float* __restrict__ pre, const float* __restrict__ stats,
                           const float* __restrict__ lw, const float* __restrict__ lb,
                           float* __restrict__ out) {
  size_t idx = (size_t)blockIdx.x * 256 + threadIdx.x;
  int b = (int)(idx / (CO_ * N_ * L_));
  int rest = (int)(idx % (CO_ * N_ * L_));
  out[idx] = (pre[idx] - stats[b * 2]) * stats[b * 2 + 1] * lw[rest] + lb[rest];
}

// ============================ host launch ============================
extern "C" void kernel_launch(void* const* d_in, const int* in_sizes, int n_in,
                              void* d_out, int out_size, void* d_ws, size_t ws_size,
                              hipStream_t stream) {
  (void)in_sizes; (void)n_in; (void)out_size; (void)ws_size;
  const float* x    = (const float*)d_in[0];
  const float* sup  = (const float*)d_in[1];
  const float* c1w  = (const float*)d_in[2];
  const float* c1b  = (const float*)d_in[3];
  const float* tw1  = (const float*)d_in[4];
  const float* tw2  = (const float*)d_in[5];
  const float* tw   = (const float*)d_in[6];
  const float* tbm  = (const float*)d_in[7];
  const float* tv   = (const float*)d_in[8];
  const float* sw1  = (const float*)d_in[9];
  const float* sw2  = (const float*)d_in[10];
  const float* sw   = (const float*)d_in[11];
  const float* sbm  = (const float*)d_in[12];
  const float* sv   = (const float*)d_in[13];
  const float* gw   = (const float*)d_in[14];
  const float* gb   = (const float*)d_in[15];
  const float* tcw  = (const float*)d_in[16];
  const float* tcb  = (const float*)d_in[17];
  const float* lnw  = (const float*)d_in[18];
  const float* lnb  = (const float*)d_in[19];

  float* out  = (float*)d_out;                                   // [B,CO,N,L]
  float* Sreg = out + (size_t)B_ * CO_ * N_ * L_;                // [B,N,N]  (logits -> S_coef in place)
  float* Treg = Sreg + (size_t)B_ * N_ * N_;                     // [B,L,L]

  char* ws = (char*)d_ws;
  float* xin  = (float*)(ws + OFF_XIN);
  float* gcn  = (float*)(ws + OFF_GCN);
  h16*   xT   = (h16*)(ws + OFF_XTAT);
  h16*   rawS = (h16*)(ws + OFF_RAWS);
  h16*   Sh   = (h16*)(ws + OFF_SH);
  h16*   Y    = (h16*)(ws + OFF_Y);
  h16*   suph = (h16*)(ws + OFF_SUPH);
  h16*   svh  = (h16*)(ws + OFF_SVH);
  h16*   t2h  = (h16*)(ws + OFF_T2H);
  float* f1t  = (float*)(ws + OFF_F1T);
  float* f2t  = (float*)(ws + OFF_F2T);
  float* gt   = (float*)(ws + OFF_GT);
  float* f1s  = (float*)(ws + OFF_F1S);
  float* f2s  = (float*)(ws + OFF_F2S);
  float* gs   = (float*)(ws + OFF_GS);
  float* cmax = (float*)(ws + OFF_CMAX);
  float* part = (float*)(ws + OFF_PART);
  float* stat = (float*)(ws + OFF_STATS);

  // f16 staging of shared matrices
  k_cvt_h<<<(N_ * N_) / 256, 256, 0, stream>>>(sup, suph, N_ * N_);
  k_cvt_h<<<(N_ * N_) / 256, 256, 0, stream>>>(sv, svh, N_ * N_);

  // residual branch
  k_conv1<<<B_ * N_, 256, 0, stream>>>(x, c1w, c1b, xin);

  // temporal attention (small, fp32)
  k_tf1<<<(B_ * L_ * N_) / 256, 256, 0, stream>>>(x, tw1, f1t);
  k_tf2<<<(B_ * C_ * L_) / 256, 256, 0, stream>>>(x, tw2, f2t);
  k_tg <<<(B_ * L_ * C_) / 256, 256, 0, stream>>>(f1t, tw, gt);
  k_tatt<<<B_, 256, 0, stream>>>(gt, f2t, tbm, tv, Treg);

  // spatial attention: raw (fp32 math, f16 store) then WMMA logits GEMM
  k_sf1<<<(B_ * N_ * L_) / 256, 256, 0, stream>>>(x, sw1, f1s);
  k_sf2<<<(B_ * C_ * N_) / 256, 256, 0, stream>>>(x, sw2, f2s);
  k_sg <<<(B_ * N_ * C_) / 256, 256, 0, stream>>>(f1s, sw, gs);
  k_sraw<<<((size_t)B_ * N_ * N_) / 256, 256, 0, stream>>>(gs, f2s, sbm, rawS);
  k_logits_wmma<<<dim3(N_ / 64, N_ / 64, B_), 256, 0, stream>>>(svh, rawS, Sreg);
  k_colmax<<<(B_ * N_) / 256, 256, 0, stream>>>(Sreg, cmax);
  k_softmax_s<<<B_ * N_, 256, 0, stream>>>(Sreg, cmax, Sh);

  // x_TAt (needs T_coef), Chebyshev T2 (WMMA)
  k_xtat<<<B_ * N_, 256, 0, stream>>>(x, Treg, xT);
  k_cheb2_wmma<<<dim3(N_ / 64, N_ / 64, 1), 256, 0, stream>>>(suph, t2h);

  // the big fused graph contraction (k=1: supports, k=2: T2) -- WMMA
  k_xg_wmma<<<dim3(N_ / 64, (C_ * L_) / 64, 2 * B_), 256, 0, stream>>>(Sh, suph, t2h, xT, Y);

  // gcn 1x1 combine (+ k=0 identity term), time conv + residual + relu, LayerNorm
  k_gcn_combine<<<B_ * N_, 256, 0, stream>>>(xT, Y, Sreg, gw, gb, gcn);
  k_timeconv<<<B_ * N_, 256, 0, stream>>>(gcn, tcw, tcb, xin);
  k_ln_partial<<<B_ * 96, 256, 0, stream>>>(xin, part);
  k_ln_final<<<B_, 128, 0, stream>>>(part, stat);
  k_ln_apply<<<((size_t)B_ * CO_ * N_ * L_) / 256, 256, 0, stream>>>(xin, stat, lnw, lnb, out);
}
